// GNN_node_81621558493500
// MI455X (gfx1250) — matrix operand order repaired and checked by reading, not compiled, not run on hardware
//
#include <hip/hip_runtime.h>
#include <hip/hip_bf16.h>

#define EMB 64
#define NUM_LAYERS 4
#define ATOM_VOCAB 64
#define BOND_VOCAB 8

typedef _Float16 v16h __attribute__((ext_vector_type(16)));
typedef float    v8f  __attribute__((ext_vector_type(8)));

// ---------------------------------------------------------------------------
// AtomEncoder: h[n,:] = sum_{c<9} atom_emb[c][x[n,c]][:]   (float2 per thread)
// ---------------------------------------------------------------------------
__global__ void atom_encoder_kernel(const int* __restrict__ x,
                                    const float* __restrict__ atom_emb,
                                    float* __restrict__ h, int N) {
    int tid = blockIdx.x * blockDim.x + threadIdx.x;
    if (tid >= N * 32) return;
    int n = tid >> 5;
    int d = (tid & 31) * 2;
    const int* xr = x + n * 9;
    float ax = 0.f, ay = 0.f;
#pragma unroll
    for (int c = 0; c < 9; ++c) {
        const float* e = atom_emb + ((size_t)(c * ATOM_VOCAB + xr[c])) * EMB + d;
        ax += e[0];
        ay += e[1];
    }
    h[(size_t)n * EMB + d]     = ax;
    h[(size_t)n * EMB + d + 1] = ay;
}

// batch (int) -> float tail of d_out
__global__ void batch_copy_kernel(const int* __restrict__ batch,
                                  float* __restrict__ out, int N) {
    int i = blockIdx.x * blockDim.x + threadIdx.x;
    if (i < N) out[i] = (float)batch[i];
}

// agg = (1 + eps[layer]) * h
__global__ void init_agg_kernel(const float* __restrict__ h,
                                const float* __restrict__ eps, int layer,
                                float* __restrict__ agg, int total) {
    int i = blockIdx.x * blockDim.x + threadIdx.x;
    if (i >= total) return;
    agg[i] = (1.0f + eps[layer]) * h[i];
}

// ---------------------------------------------------------------------------
// Edge phase: msg = relu(h[src] + bond_emb(edge_attr)); scatter-add at dst.
// One 32-lane wave per edge, float2 per lane, global f32 atomics (L2 resident).
// ---------------------------------------------------------------------------
__global__ void edge_msg_kernel(const float* __restrict__ h,
                                const int* __restrict__ src,
                                const int* __restrict__ dst,
                                const int* __restrict__ edge_attr,
                                const float* __restrict__ bond,  // [3,8,64] for this layer
                                float* __restrict__ agg, int E) {
    int tid = blockIdx.x * blockDim.x + threadIdx.x;
    int e = tid >> 5;
    if (e >= E) return;
    int d  = (tid & 31) * 2;
    int a0 = edge_attr[e * 3 + 0];
    int a1 = edge_attr[e * 3 + 1];
    int a2 = edge_attr[e * 3 + 2];
    int s  = src[e];
    int t  = dst[e];
    const float* b0 = bond + (size_t)(0 * BOND_VOCAB + a0) * EMB + d;
    const float* b1 = bond + (size_t)(1 * BOND_VOCAB + a1) * EMB + d;
    const float* b2 = bond + (size_t)(2 * BOND_VOCAB + a2) * EMB + d;
    const float* hs = h + (size_t)s * EMB + d;
    float mx = fmaxf(hs[0] + b0[0] + b1[0] + b2[0], 0.f);
    float my = fmaxf(hs[1] + b0[1] + b1[1] + b2[1], 0.f);
    float* ap = agg + (size_t)t * EMB + d;
    atomicAdd(ap,     mx);
    atomicAdd(ap + 1, my);
}

// zero sum/sumsq accumulators (layout: sum1[128] sq1[128] sum2[64] sq2[64])
__global__ void zero_stats_kernel(float* __restrict__ stats) {
    int i = threadIdx.x;
    if (i < 384) stats[i] = 0.f;
}

// ---------------------------------------------------------------------------
// GEMM1: Z[N,128] = A[N,64] @ W[64,128] + bias, via v_wmma_f32_16x16x32_f16.
// Block = 4 waves, all sharing one 16-col tile of W staged in LDS as f16,
// transposed so each lane's B fragment is one 32B contiguous ds_load.
// ---------------------------------------------------------------------------
__global__ void gemm_lin1_wmma(const float* __restrict__ A,
                               const float* __restrict__ W,
                               const float* __restrict__ bias,
                               float* __restrict__ Z, int N) {
    __shared__ _Float16 Wht[16 * 64];  // [c][k] transposed tile, 2 KB
    const int t = threadIdx.x;
    const int colTile = blockIdx.y;    // 0..7

    // cooperative stage: 1024 elements / 128 threads
#pragma unroll
    for (int i = 0; i < 8; ++i) {
        int idx = t + i * 128;
        int k = idx >> 4;              // 0..63
        int c = idx & 15;              // 0..15
        Wht[c * 64 + k] = (_Float16)W[(size_t)k * 128 + colTile * 16 + c];
    }
    __syncthreads();

    const int lane = t & 31;
    const int wave = t >> 5;
    const int numRowTiles = (N + 15) >> 4;
    const int rowTile = blockIdx.x * 4 + wave;
    if (rowTile >= numRowTiles) return;            // wave-uniform, after barrier
    const int half = lane >> 4;
    const int r    = lane & 15;
    const int row  = rowTile * 16 + r;
    const int rc   = row < N ? row : N - 1;        // clamp: EXEC all-1s at WMMA
    const int col  = colTile * 16 + r;

    v8f acc = {};
#pragma unroll
    for (int kb = 0; kb < 2; ++kb) {
        const int k0 = kb * 32;
        v16h a;
        const float* arow = A + (size_t)rc * 64 + k0 + 16 * half;
#pragma unroll
        for (int e = 0; e < 8; ++e) a[e]     = (_Float16)arow[e];        // K = k0+16h+e
#pragma unroll
        for (int e = 0; e < 8; ++e) a[8 + e] = (_Float16)arow[8 + e];    // K = k0+16h+8+e
        // B fragment: k = k0 + 16*half + e, col = r  -> 16 contiguous halfs
        v16h b = *(const v16h*)(&Wht[r * 64 + k0 + 16 * half]);
        acc = __builtin_amdgcn_wmma_f32_16x16x32_f16(false, a, false, b,
                                                     (short)0, acc, false, false);
    }
    const float bv = bias[col];
    if (rowTile * 16 + 16 <= N) {                  // wave-uniform fast path
        float* zp = Z + (size_t)(rowTile * 16 + 8 * half) * 128 + col;
#pragma unroll
        for (int j = 0; j < 8; ++j) zp[(size_t)j * 128] = acc[j] + bv;
    } else {
#pragma unroll
        for (int j = 0; j < 8; ++j) {
            int orow = rowTile * 16 + j + 8 * half;
            if (orow < N) Z[(size_t)orow * 128 + col] = acc[j] + bv;
        }
    }
}

// ---------------------------------------------------------------------------
// GEMM2: Z2[N,64] = relu(s1*Z1 + t1)[N,128] @ W[128,64] + bias  (BN1 fused into
// the f16 A-fragment conversion), four x32 WMMA steps, W tile in LDS.
// ---------------------------------------------------------------------------
__global__ void gemm_lin2_wmma(const float* __restrict__ Z1,
                               const float* __restrict__ s1,
                               const float* __restrict__ t1,
                               const float* __restrict__ W,
                               const float* __restrict__ bias,
                               float* __restrict__ Z2, int N) {
    __shared__ _Float16 Wht[16 * 128]; // [c][k] transposed tile, 4 KB
    const int t = threadIdx.x;
    const int colTile = blockIdx.y;    // 0..3

#pragma unroll
    for (int i = 0; i < 16; ++i) {
        int idx = t + i * 128;
        int k = idx >> 4;              // 0..127
        int c = idx & 15;              // 0..15
        Wht[c * 128 + k] = (_Float16)W[(size_t)k * 64 + colTile * 16 + c];
    }
    __syncthreads();

    const int lane = t & 31;
    const int wave = t >> 5;
    const int numRowTiles = (N + 15) >> 4;
    const int rowTile = blockIdx.x * 4 + wave;
    if (rowTile >= numRowTiles) return;
    const int half = lane >> 4;
    const int r    = lane & 15;
    const int row  = rowTile * 16 + r;
    const int rc   = row < N ? row : N - 1;
    const int col  = colTile * 16 + r;
    const float* zr = Z1 + (size_t)rc * 128;

    v8f acc = {};
#pragma unroll
    for (int kb = 0; kb < 4; ++kb) {
        const int k0 = kb * 32;
        v16h a;
#pragma unroll
        for (int e = 0; e < 8; ++e) {
            int k = k0 + 16 * half + e;
            a[e] = (_Float16)fmaxf(fmaf(s1[k], zr[k], t1[k]), 0.f);
        }
#pragma unroll
        for (int e = 0; e < 8; ++e) {
            int k = k0 + 16 * half + 8 + e;
            a[8 + e] = (_Float16)fmaxf(fmaf(s1[k], zr[k], t1[k]), 0.f);
        }
        v16h b = *(const v16h*)(&Wht[r * 128 + k0 + 16 * half]);
        acc = __builtin_amdgcn_wmma_f32_16x16x32_f16(false, a, false, b,
                                                     (short)0, acc, false, false);
    }
    const float bv = bias[col];
    if (rowTile * 16 + 16 <= N) {
        float* zp = Z2 + (size_t)(rowTile * 16 + 8 * half) * 64 + col;
#pragma unroll
        for (int j = 0; j < 8; ++j) zp[(size_t)j * 64] = acc[j] + bv;
    } else {
#pragma unroll
        for (int j = 0; j < 8; ++j) {
            int orow = rowTile * 16 + j + 8 * half;
            if (orow < N) Z2[(size_t)orow * 64 + col] = acc[j] + bv;
        }
    }
}

// ---------------------------------------------------------------------------
// Per-column sum / sumsq over [N,C] (C = 64 or 128). Coalesced: 256 threads =
// (256/C) row-groups x C columns; private accumulate, LDS combine, global atomic.
// ---------------------------------------------------------------------------
__global__ void col_stats_kernel(const float* __restrict__ Z, int N, int C,
                                 float* __restrict__ sum, float* __restrict__ sq) {
    __shared__ float ls[256];
    __shared__ float lq[256];
    const int t = threadIdx.x;
    const int G = 256 / C;
    const int g = t / C;
    const int c = t - g * C;
    float s = 0.f, q = 0.f;
    for (int row = blockIdx.x * G + g; row < N; row += gridDim.x * G) {
        float v = Z[(size_t)row * C + c];
        s += v;
        q += v * v;
    }
    ls[t] = s;
    lq[t] = q;
    __syncthreads();
    if (t < C) {
        float S = 0.f, Q = 0.f;
        for (int gg = 0; gg < G; ++gg) { S += ls[gg * C + t]; Q += lq[gg * C + t]; }
        atomicAdd(&sum[t], S);
        atomicAdd(&sq[t], Q);
    }
}

// BN scale/shift: s = gamma*rsqrt(var+eps), t = beta - mean*s  (population var)
__global__ void finalize_bn_kernel(const float* __restrict__ sum,
                                   const float* __restrict__ sq,
                                   const float* __restrict__ gamma,
                                   const float* __restrict__ beta,
                                   float* __restrict__ s, float* __restrict__ t,
                                   int N, int C) {
    int c = blockIdx.x * blockDim.x + threadIdx.x;
    if (c >= C) return;
    float invN = 1.0f / (float)N;
    float mean = sum[c] * invN;
    float var  = sq[c] * invN - mean * mean;
    float rstd = rsqrtf(var + 1e-5f);
    float sc   = gamma[c] * rstd;
    s[c] = sc;
    t[c] = beta[c] - mean * sc;
}

// h = s2*Z2 + t2  (ReLU on all but last layer)
__global__ void bn_apply_kernel(const float* __restrict__ Z,
                                const float* __restrict__ s,
                                const float* __restrict__ t,
                                float* __restrict__ H, int total, int doRelu) {
    int i = blockIdx.x * blockDim.x + threadIdx.x;
    if (i >= total) return;
    int c = i & 63;
    float v = fmaf(s[c], Z[i], t[c]);
    if (doRelu) v = fmaxf(v, 0.f);
    H[i] = v;
}

// ---------------------------------------------------------------------------
extern "C" void kernel_launch(void* const* d_in, const int* in_sizes, int n_in,
                              void* d_out, int out_size, void* d_ws, size_t ws_size,
                              hipStream_t stream) {
    (void)n_in; (void)out_size; (void)ws_size;
    const int*   x         = (const int*)d_in[0];
    const int*   edgeIndex = (const int*)d_in[1];
    const int*   edgeAttr  = (const int*)d_in[2];
    const int*   batch     = (const int*)d_in[3];
    const float* atomEmb   = (const float*)d_in[4];
    const float* bondEmb   = (const float*)d_in[5];
    const float* eps       = (const float*)d_in[6];
    const float* W1        = (const float*)d_in[7];
    const float* b1        = (const float*)d_in[8];
    const float* bn1g      = (const float*)d_in[9];
    const float* bn1b      = (const float*)d_in[10];
    const float* W2        = (const float*)d_in[11];
    const float* b2        = (const float*)d_in[12];
    const float* bng       = (const float*)d_in[13];
    const float* bnb       = (const float*)d_in[14];

    const int N = in_sizes[0] / 9;
    const int E = in_sizes[1] / 2;
    const int* src = edgeIndex;
    const int* dst = edgeIndex + E;

    float* h        = (float*)d_out;              // [N,64] final node features
    float* outBatch = h + (size_t)N * 64;         // [N] batch ids as float

    float* zA    = (float*)d_ws;                  // [N,64]  agg / z2
    float* z1    = zA + (size_t)N * 64;           // [N,128] hidden activations
    float* stats = z1 + (size_t)N * 128;
    float* sum1 = stats,       *sq1 = stats + 128;
    float* sum2 = stats + 256, *sq2 = stats + 320;
    float* s1   = stats + 384, *t1  = stats + 512;
    float* s2   = stats + 640, *t2  = stats + 704;

    atom_encoder_kernel<<<(N * 32 + 255) / 256, 256, 0, stream>>>(x, atomEmb, h, N);
    batch_copy_kernel<<<(N + 255) / 256, 256, 0, stream>>>(batch, outBatch, N);

    const int rowTiles = (N + 15) / 16;
    dim3 grid1((rowTiles + 3) / 4, 8);   // 128 output cols
    dim3 grid2((rowTiles + 3) / 4, 4);   // 64 output cols

    for (int layer = 0; layer < NUM_LAYERS; ++layer) {
        zero_stats_kernel<<<1, 384, 0, stream>>>(stats);
        init_agg_kernel<<<(N * 64 + 255) / 256, 256, 0, stream>>>(h, eps, layer, zA, N * 64);
        edge_msg_kernel<<<(E * 32 + 255) / 256, 256, 0, stream>>>(
            h, src, dst, edgeAttr,
            bondEmb + (size_t)layer * 3 * BOND_VOCAB * EMB, zA, E);
        gemm_lin1_wmma<<<grid1, 128, 0, stream>>>(
            zA, W1 + (size_t)layer * 64 * 128, b1 + (size_t)layer * 128, z1, N);
        col_stats_kernel<<<256, 256, 0, stream>>>(z1, N, 128, sum1, sq1);
        finalize_bn_kernel<<<1, 128, 0, stream>>>(sum1, sq1,
            bn1g + (size_t)layer * 128, bn1b + (size_t)layer * 128, s1, t1, N, 128);
        gemm_lin2_wmma<<<grid2, 128, 0, stream>>>(
            z1, s1, t1, W2 + (size_t)layer * 128 * 64, b2 + (size_t)layer * 64, zA, N);
        col_stats_kernel<<<256, 256, 0, stream>>>(zA, N, 64, sum2, sq2);
        finalize_bn_kernel<<<1, 64, 0, stream>>>(sum2, sq2,
            bng + (size_t)layer * 64, bnb + (size_t)layer * 64, s2, t2, N, 64);
        bn_apply_kernel<<<(N * 64 + 255) / 256, 256, 0, stream>>>(
            zA, s2, t2, h, N * 64, layer < NUM_LAYERS - 1 ? 1 : 0);
    }
}
